// EdgePredictor_2018634629213
// MI455X (gfx1250) — matrix operand order repaired and checked by reading, not compiled, not run on hardware
//
#include <hip/hip_runtime.h>
#include <hip/hip_bf16.h>
#include <math.h>

typedef __attribute__((ext_vector_type(16))) _Float16 v16h;
typedef __attribute__((ext_vector_type(8)))  _Float16 v8h;
typedef __attribute__((ext_vector_type(8)))  float    v8f;

#define NB 8
#define NN 256
#define DD 64
#define HH 256
#define NPAIRS 32640   /* 256*255/2 */
#define PPB 32         /* pairs per block */
#define ROWS (PPB*2)   /* 64 activation rows per block */
#define RSTRIDE 264    /* f16 elements; 528B row stride, 16B aligned, bank-staggered */

__device__ __forceinline__ float elu_f(float v) {
    return v > 0.f ? v : (__expf(v) - 1.f);
}

// ---------------------------------------------------------------------------
// Kernel 1: per-node layer-1 halves.  A[n] = x[n]@W1[0:64]  + b1
//                                     Bv[n] = x[n]@W1[64:128]
// ---------------------------------------------------------------------------
__global__ void prep_node(const float* __restrict__ x, const float* __restrict__ W1,
                          const float* __restrict__ b1, float* __restrict__ Ap,
                          float* __restrict__ Bv) {
    __shared__ float xs[DD];
    const int n = blockIdx.x, b = blockIdx.y;
    const int h = threadIdx.x;
    if (h < DD) xs[h] = x[((size_t)b * NN + n) * DD + h];
    __syncthreads();
    float accA = b1[h], accB = 0.f;
    #pragma unroll 8
    for (int d = 0; d < DD; ++d) {
        const float xv = xs[d];
        accA = fmaf(xv, W1[d * HH + h], accA);
        accB = fmaf(xv, W1[(DD + d) * HH + h], accB);
    }
    const size_t o = ((size_t)b * NN + n) * HH + h;
    Ap[o] = accA;
    Bv[o] = accB;
}

// ---------------------------------------------------------------------------
// Kernel 2: pack W2 (f32 256x256, KxN) into f16 WMMA B-fragment order.
// Fragment (nt,kc): 32 lanes x 16 f16.  Lane (n=lane&15, hi=lane>>4),
// element 2j   = W2[kc*32 + hi*16 + 2j    ][nt*16+n]
// element 2j+1 = W2[kc*32 + hi*16 + 2j + 1][nt*16+n]
// ---------------------------------------------------------------------------
__global__ void pack_w2(const float* __restrict__ W2, _Float16* __restrict__ pw2) {
    const int idx  = blockIdx.x * 256 + threadIdx.x;  // 4096 = 16 nt * 8 kc * 32 lanes
    const int nt   = idx >> 8;
    const int rem  = idx & 255;
    const int kc   = rem >> 5;
    const int lane = rem & 31;
    const int n    = lane & 15, hi = lane >> 4;
    _Float16* dst = pw2 + ((size_t)(nt * 8 + kc) * 32 + lane) * 16;
    #pragma unroll
    for (int j = 0; j < 8; ++j) {
        const int k0 = kc * 32 + hi * 16 + 2 * j;
        dst[2 * j]     = (_Float16)W2[(size_t)k0 * HH + nt * 16 + n];
        dst[2 * j + 1] = (_Float16)W2[(size_t)(k0 + 1) * HH + nt * 16 + n];
    }
}

// ---------------------------------------------------------------------------
// Kernel 3: zero the (poisoned) diagonal of the output adjacency.
// ---------------------------------------------------------------------------
__global__ void zero_diag(float* __restrict__ out) {
    const int idx = blockIdx.x * 256 + threadIdx.x;  // 2048 = NB*NN
    const int b = idx >> 8, n = idx & 255;
    out[((size_t)b * NN + n) * NN + n] = 0.f;
}

// ---------------------------------------------------------------------------
// Kernel 4: main.  32 pairs/block -> 64 rows.  h1 (LDS f16) @ W2 via WMMA,
// epilogue: +b2, ELU, dot with W3, lane-reduce, sigmoid of mean, scatter.
// ---------------------------------------------------------------------------
__global__ __launch_bounds__(256) void edge_main(
    const float* __restrict__ Ap, const float* __restrict__ Bv,
    const _Float16* __restrict__ pw2, const float* __restrict__ b2,
    const float* __restrict__ W3, const float* __restrict__ b3,
    float* __restrict__ out)
{
    __shared__ int pj[PPB], pi[PPB];
    __shared__ float logitAcc[ROWS];
    __shared__ alignas(16) _Float16 h1[ROWS * RSTRIDE];

    const int tid = threadIdx.x;
    const int bb  = blockIdx.y;
    const int m0  = blockIdx.x * PPB;

    // --- pair index decode (triu order, k=1) + logit accumulator init ---
    if (tid < PPB) {
        const int m = m0 + tid;
        const float a = 2.f * (NN - 1) + 1.f;
        int j = (int)floorf((a - sqrtf(a * a - 8.f * (float)m)) * 0.5f);
        if (j < 0) j = 0;
        if (j > NN - 2) j = NN - 2;
        while (j < NN - 2 && ((j + 1) * (NN - 1) - ((j + 1) * j) / 2) <= m) ++j;
        while (j > 0 && (j * (NN - 1) - (j * (j - 1)) / 2) > m) --j;
        const int base = j * (NN - 1) - (j * (j - 1)) / 2;
        pj[tid] = j;
        pi[tid] = j + 1 + (m - base);
    }
    if (tid < ROWS) logitAcc[tid] = 0.f;
    __syncthreads();

    // --- build h1 = elu(A[src1] + Bv[src2]) in f16, rows 2p+t ---
    {
        const int r  = tid >> 2;          // 0..63
        const int c0 = (tid & 3) * 64;
        const int p = r >> 1, t = r & 1;
        const int j = pj[p], i = pi[p];
        const int na = t ? i : j;         // feeds W1a half (+b1)
        const int nb = t ? j : i;         // feeds W1b half
        const float* s1 = Ap + ((size_t)bb * NN + na) * HH + c0;
        const float* s2 = Bv + ((size_t)bb * NN + nb) * HH + c0;
        _Float16* dst = h1 + r * RSTRIDE + c0;
        #pragma unroll 8
        for (int c = 0; c < 64; ++c) {
            dst[c] = (_Float16)elu_f(s1[c] + s2[c]);
        }
    }
    __syncthreads();

    // --- layer-2 GEMM: 64x256 @ 256x256, f16 WMMA, f32 accumulate ---
    const int w    = tid >> 5;            // wave 0..7 -> N-tiles {2w, 2w+1}
    const int lane = tid & 31;
    const int nlo  = lane & 15;
    const int hi8  = (lane >> 4) * 8;

    v8f acc[2][4];
    {
        const v8f z = {};
        #pragma unroll
        for (int s = 0; s < 2; ++s)
            #pragma unroll
            for (int rg = 0; rg < 4; ++rg) acc[s][rg] = z;
    }

    for (int kc = 0; kc < 8; ++kc) {
        v16h afr[4];
        #pragma unroll
        for (int rg = 0; rg < 4; ++rg) {
            // A in contiguous k order (same as packed W2): lanes 0-15 -> K 0..15, lanes 16-31 -> K 16..31
            const _Float16* ap = h1 + (rg * 16 + nlo) * RSTRIDE + kc * 32 + (lane >> 4) * 16;
            const v8h a0 = *(const v8h*)ap;
            const v8h a1 = *(const v8h*)(ap + 8);
            afr[rg] = __builtin_shufflevector(a0, a1,
                0, 1, 2, 3, 4, 5, 6, 7, 8, 9, 10, 11, 12, 13, 14, 15);
        }
        #pragma unroll
        for (int s = 0; s < 2; ++s) {
            const int nt = w * 2 + s;
            const v16h bfr = *(const v16h*)(pw2 + ((size_t)(nt * 8 + kc) * 32 + lane) * 16);
            #pragma unroll
            for (int rg = 0; rg < 4; ++rg) {
                acc[s][rg] = __builtin_amdgcn_wmma_f32_16x16x32_f16(
                    false, afr[rg], false, bfr, (short)0, acc[s][rg], false, false);
            }
        }
    }

    // --- epilogue: +b2, elu, * W3 column, reduce across 16-lane halves ---
    float partial[4][8];
    #pragma unroll
    for (int rg = 0; rg < 4; ++rg)
        #pragma unroll
        for (int r = 0; r < 8; ++r) partial[rg][r] = 0.f;

    #pragma unroll
    for (int s = 0; s < 2; ++s) {
        const int ncol = (w * 2 + s) * 16 + nlo;
        const float b2v = b2[ncol];
        const float w3v = W3[ncol];
        #pragma unroll
        for (int rg = 0; rg < 4; ++rg) {
            #pragma unroll
            for (int r = 0; r < 8; ++r) {
                const float hv = elu_f(acc[s][rg][r] + b2v);
                partial[rg][r] = fmaf(hv, w3v, partial[rg][r]);
            }
        }
    }
    #pragma unroll
    for (int rg = 0; rg < 4; ++rg) {
        #pragma unroll
        for (int r = 0; r < 8; ++r) {
            float v = partial[rg][r];
            v += __shfl_xor(v, 1, 32);
            v += __shfl_xor(v, 2, 32);
            v += __shfl_xor(v, 4, 32);
            v += __shfl_xor(v, 8, 32);
            if (nlo == 0) atomicAdd(&logitAcc[rg * 16 + hi8 + r], v);
        }
    }
    __syncthreads();

    // --- sigmoid of mean logit, scatter symmetric entries ---
    if (tid < PPB) {
        const float l = 0.5f * (logitAcc[2 * tid] + logitAcc[2 * tid + 1]) + b3[0];
        const float prob = 1.f / (1.f + __expf(-l));
        const int j = pj[tid], i = pi[tid];
        out[((size_t)bb * NN + j) * NN + i] = prob;
        out[((size_t)bb * NN + i) * NN + j] = prob;
    }
}

// ---------------------------------------------------------------------------
extern "C" void kernel_launch(void* const* d_in, const int* in_sizes, int n_in,
                              void* d_out, int out_size, void* d_ws, size_t ws_size,
                              hipStream_t stream) {
    const float* x  = (const float*)d_in[0];
    const float* W1 = (const float*)d_in[1];
    const float* b1 = (const float*)d_in[2];
    const float* W2 = (const float*)d_in[3];
    const float* b2 = (const float*)d_in[4];
    const float* W3 = (const float*)d_in[5];
    const float* b3 = (const float*)d_in[6];
    float* out = (float*)d_out;

    float*    Ap  = (float*)d_ws;                       // 8*256*256 f32 = 2 MB
    float*    Bv  = Ap + (size_t)NB * NN * HH;          // 2 MB
    _Float16* pw2 = (_Float16*)(Bv + (size_t)NB * NN * HH);  // 128 KB packed f16 W2

    prep_node<<<dim3(NN, NB), HH, 0, stream>>>(x, W1, b1, Ap, Bv);
    pack_w2<<<16, 256, 0, stream>>>(W2, pw2);
    zero_diag<<<(NB * NN) / 256, 256, 0, stream>>>(out);
    edge_main<<<dim3(NPAIRS / PPB, NB), 256, 0, stream>>>(Ap, Bv, pw2, b2, W3, b3, out);
}
